// Rcnet_52527450030482
// MI455X (gfx1250) — matrix operand order repaired and checked by reading, not compiled, not run on hardware
//
#include <hip/hip_runtime.h>

// ---------------------------------------------------------------------------
// CDNA5 (gfx1250) RC-Net forward.
// Matrix math (3x3/1x1 convs as implicit GEMM, attention einsums) runs on
// v_wmma_f32_16x16x32_f16 with 4-way M register blocking: each wave computes
// a 64x16 output strip, loading each activation (B) fragment once per K-step
// and reusing it across 4 weight (A) fragments -> 4 WMMAs. Conv weights are
// device-repacked to [rs][CoutP][CinP] f16 (zero padded) so A fragments are
// contiguous b128 loads and the WMMA K dimension is pure channels. GEMMs are
// specialized per transpose combination so contiguous-K operands use 128-bit
// loads. Elementwise ops (BN/ReLU fused into conv epilogue, pool, pixel
// merge/unmerge, bilinear up2, depthwise 3x3, softmax) are VALU kernels.
// ---------------------------------------------------------------------------

typedef __attribute__((ext_vector_type(16))) _Float16 v16h;
typedef __attribute__((ext_vector_type(8)))  _Float16 v8h;
typedef __attribute__((ext_vector_type(8)))  float    v8f;
typedef __attribute__((ext_vector_type(4)))  float    v4f;

__device__ __forceinline__ v16h cat8h(v8h lo, v8h hi) {
    return __builtin_shufflevector(lo, hi,
               0, 1, 2, 3, 4, 5, 6, 7, 8, 9, 10, 11, 12, 13, 14, 15);
}

// ---------------- WMMA implicit-GEMM conv (ks = 1 or 3, stride 1) ----------
// Weights pre-repacked f16: w[((rs*CoutP)+co)*CinP + ci], zero padded.
// MT = M-tiles per wave (1 or 4); B fragment shared across the MT WMMAs.
template <int MT>
__global__ void k_conv_wmma(const float* __restrict__ x,
                            const _Float16* __restrict__ w,
                            const float* __restrict__ scale,
                            const float* __restrict__ shift,
                            float* __restrict__ out,
                            int Bn, int Cin, int H, int W, int Cout,
                            int ks, int pad, int relu, int CinP, int CoutP)
{
    int wid  = (blockIdx.x * blockDim.x + threadIdx.x) >> 5;
    int lane = threadIdx.x & 31;
    int HW   = H * W;
    int ntN  = HW >> 4;                  // HW is a multiple of 16 everywhere
    int ntM  = CoutP / (16 * MT);
    int total = Bn * ntM * ntN;
    if (wid >= total) return;
    int nt = wid % ntN;
    int mt = (wid / ntN) % ntM;
    int b  = wid / (ntN * ntM);

    int hi   = lane >> 4;
    int aRow = mt * (16 * MT) + (lane & 15);   // first A row of the strip
    int p    = nt * 16 + (lane & 15);          // pixel (B column)
    int ph = p / W, pw = p - (p / W) * W;

    const float* xb = x + (size_t)b * Cin * HW;
    int ksq = ks * ks;
    v8f acc[MT];
#pragma unroll
    for (int t = 0; t < MT; ++t) acc[t] = (v8f){};

    for (int rs = 0; rs < ksq; ++rs) {
        int dh = rs / ks - pad, dw = rs - (rs / ks) * ks - pad;
        int hh = ph + dh, ww = pw + dw;
        bool pv = ((unsigned)hh < (unsigned)H) && ((unsigned)ww < (unsigned)W);
        const float*    xrs = xb + ((size_t)hh * W + ww);
        const _Float16* wrs = w + ((size_t)rs * CoutP + aRow) * CinP;

        for (int kk = 0; kk < Cin; kk += 32) {
            // ---- B: 16 channel-strided gathers, shared by all MT WMMAs ----
            v16h bv;
            int ci0 = kk + (hi << 4);
            if (pv && (ci0 + 16) <= Cin) {
                const float* bp = xrs + (size_t)ci0 * HW;
#pragma unroll
                for (int j = 0; j < 16; ++j) { bv[j] = (_Float16)bp[0]; bp += HW; }
            } else {
#pragma unroll
                for (int j = 0; j < 16; ++j) {
                    int ci = ci0 + j;
                    float f = (pv && ci < Cin) ? xrs[(size_t)ci * HW] : 0.f;
                    bv[j] = (_Float16)f;
                }
            }
            // ---- A: MT fragments, each two contiguous b128 loads ----
#pragma unroll
            for (int t = 0; t < MT; ++t) {
                const _Float16* wr = wrs + (size_t)(16 * t) * CinP + kk;
                v8h alo = *(const v8h*)(wr + (hi << 4));
                v8h ahi = *(const v8h*)(wr + 8 + (hi << 4));
                acc[t] = __builtin_amdgcn_wmma_f32_16x16x32_f16(
                             false, cat8h(alo, ahi), false, bv,
                             (short)0, acc[t], false, false);
            }
        }
    }

    // epilogue: fused bias+BN (+ReLU)
    float* ob = out + (size_t)b * Cout * HW;
    int n = nt * 16 + (lane & 15);
#pragma unroll
    for (int t = 0; t < MT; ++t) {
#pragma unroll
        for (int r = 0; r < 8; ++r) {
            int m = mt * (16 * MT) + 16 * t + r + (hi << 3);
            if (m < Cout) {
                float v = acc[t][r];
                float sc = scale ? scale[m] : 1.f;
                float sh = shift ? shift[m] : 0.f;
                v = v * sc + sh;
                if (relu) v = fmaxf(v, 0.f);
                ob[(size_t)m * HW + n] = v;
            }
        }
    }
}

// ---------------- generic batched WMMA GEMM (attention einsums) ------------
// C[b,m,n] = sum_k A(m,k)*B(k,n), 64x16 strip per wave (4-way M blocking).
// A(m,k) = TA ? A[k*lda+m] : A[m*lda+k];  B(k,n) = TB ? B[n*ldb+k] : B[k*ldb+n]
// All M are multiples of 64, N of 16, K of 32 at the call sites.
template <int TA, int TB>
__global__ void k_gemm_wmma(const float* __restrict__ A,
                            const float* __restrict__ Bm,
                            float* __restrict__ C,
                            int M, int N, int Kd,
                            int lda, int ldb, int ldc,
                            long long sA, long long sB, long long sC, int batch)
{
    int wid  = (blockIdx.x * blockDim.x + threadIdx.x) >> 5;
    int lane = threadIdx.x & 31;
    int ntN = N >> 4, ntM = M >> 6;
    int total = batch * ntM * ntN;
    if (wid >= total) return;
    int nt = wid % ntN;
    int mt = (wid / ntN) % ntM;
    int b  = wid / (ntN * ntM);

    const float* Ab = A  + (size_t)b * sA;
    const float* Bb = Bm + (size_t)b * sB;
    float*       Cb = C  + (size_t)b * sC;

    int hi = lane >> 4;
    int aRow = mt * 64 + (lane & 15);
    int bCol = nt * 16 + (lane & 15);
    v8f acc[4];
#pragma unroll
    for (int t = 0; t < 4; ++t) acc[t] = (v8f){};

    for (int kk = 0; kk < Kd; kk += 32) {
        // ---- B fragment, shared across the 4 WMMAs ----
        v16h bv;
        if constexpr (TB == 1) {
            const float* bp = Bb + (size_t)bCol * ldb + kk + (hi << 3);
            v4f q0 = *(const v4f*)(bp);
            v4f q1 = *(const v4f*)(bp + 4);
            v4f q2 = *(const v4f*)(bp + 16);
            v4f q3 = *(const v4f*)(bp + 20);
#pragma unroll
            for (int j = 0; j < 4; ++j) {
                bv[j]      = (_Float16)q0[j];
                bv[j + 4]  = (_Float16)q1[j];
                bv[j + 8]  = (_Float16)q2[j];
                bv[j + 12] = (_Float16)q3[j];
            }
        } else {
#pragma unroll
            for (int j = 0; j < 16; ++j) {
                int kB = kk + ((j >> 3) << 4) + (hi << 3) + (j & 7);
                bv[j] = (_Float16)Bb[(size_t)kB * ldb + bCol];
            }
        }
        // ---- 4 A fragments ----
#pragma unroll
        for (int t = 0; t < 4; ++t) {
            v16h a;
            int row = aRow + 16 * t;
            if constexpr (TA == 0) {
                const float* ap = Ab + (size_t)row * lda + kk + (hi << 3);
                v4f q0 = *(const v4f*)(ap);
                v4f q1 = *(const v4f*)(ap + 4);
                v4f q2 = *(const v4f*)(ap + 16);
                v4f q3 = *(const v4f*)(ap + 20);
#pragma unroll
                for (int j = 0; j < 4; ++j) {
                    a[j]      = (_Float16)q0[j];
                    a[j + 4]  = (_Float16)q1[j];
                    a[j + 8]  = (_Float16)q2[j];
                    a[j + 12] = (_Float16)q3[j];
                }
            } else {
#pragma unroll
                for (int j = 0; j < 16; ++j) {
                    int kA = kk + ((j >> 3) << 4) + (hi << 3) + (j & 7);
                    a[j] = (_Float16)Ab[(size_t)kA * lda + row];
                }
            }
            acc[t] = __builtin_amdgcn_wmma_f32_16x16x32_f16(
                         false, a, false, bv, (short)0, acc[t], false, false);
        }
    }
    int n = nt * 16 + (lane & 15);
#pragma unroll
    for (int t = 0; t < 4; ++t)
#pragma unroll
        for (int r = 0; r < 8; ++r) {
            int m = mt * 64 + 16 * t + r + (hi << 3);
            Cb[(size_t)m * ldc + n] = acc[t][r];
        }
}

// ------------------------------ helpers ------------------------------------
// OIHW f32 -> [rs][CoutP][CinP] f16, zero padded
__global__ void k_repack_w(const float* __restrict__ src, _Float16* __restrict__ dst,
                           int Cout, int Cin, int ks, int CoutP, int CinP) {
    int ksq = ks * ks;
    int total = ksq * CoutP * CinP;
    int i = blockIdx.x * blockDim.x + threadIdx.x;
    if (i >= total) return;
    int ci = i % CinP;
    int co = (i / CinP) % CoutP;
    int rs = i / (CinP * CoutP);
    float v = 0.f;
    if (co < Cout && ci < Cin) v = src[((size_t)co * Cin + ci) * ksq + rs];
    dst[i] = (_Float16)v;
}

// scale = g*rsqrt(v+eps); shift = b + scale*(conv_bias - m)
__global__ void k_bnprep(const float* g, const float* b, const float* m,
                         const float* v, const float* cb,
                         float* scale, float* shift, int C) {
    int i = blockIdx.x * blockDim.x + threadIdx.x;
    if (i < C) {
        float s = g[i] * rsqrtf(v[i] + 1e-5f);
        scale[i] = s;
        float c0 = cb ? cb[i] : 0.f;
        shift[i] = b[i] + s * (c0 - m[i]);
    }
}

__global__ void k_dwconv3(const float* __restrict__ x, const float* __restrict__ w,
                          const float* __restrict__ bias, float* __restrict__ out,
                          int Bn, int C, int H, int W) {
    int i = blockIdx.x * blockDim.x + threadIdx.x;
    int total = Bn * C * H * W;
    if (i >= total) return;
    int ww = i % W, hh = (i / W) % H, c = (i / (W * H)) % C, b = i / (W * H * C);
    const float* xp = x + ((size_t)b * C + c) * H * W;
    const float* wp = w + c * 9;
    float acc = bias ? bias[c] : 0.f;
#pragma unroll
    for (int r = 0; r < 3; ++r)
#pragma unroll
        for (int s = 0; s < 3; ++s) {
            int h2 = hh + r - 1, w2 = ww + s - 1;
            if ((unsigned)h2 < (unsigned)H && (unsigned)w2 < (unsigned)W)
                acc += wp[r * 3 + s] * xp[h2 * W + w2];
        }
    out[i] = acc;
}

__global__ void k_pool2(const float* __restrict__ x, float* __restrict__ y,
                        int Bn, int C, int H, int W) {
    int Ho = H >> 1, Wo = W >> 1;
    int i = blockIdx.x * blockDim.x + threadIdx.x;
    int total = Bn * C * Ho * Wo;
    if (i >= total) return;
    int wo = i % Wo, ho = (i / Wo) % Ho, c = (i / (Wo * Ho)) % C, b = i / (Wo * Ho * C);
    const float* xp = x + (((size_t)b * C + c) * H + 2 * ho) * W + 2 * wo;
    y[i] = fmaxf(fmaxf(xp[0], xp[1]), fmaxf(xp[W], xp[W + 1]));
}

// pixel-merge: (B,C,H,W) -> (B,4C,H/2,W/2), q: 0=(ee),1=(oe),2=(eo),3=(oo)
__global__ void k_pm(const float* __restrict__ x, float* __restrict__ y,
                     int Bn, int C, int H, int W) {
    int Ho = H >> 1, Wo = W >> 1;
    int i = blockIdx.x * blockDim.x + threadIdx.x;
    int total = Bn * 4 * C * Ho * Wo;
    if (i >= total) return;
    int wo = i % Wo, ho = (i / Wo) % Ho, co = (i / (Wo * Ho)) % (4 * C), b = i / (Wo * Ho * 4 * C);
    int q = co / C, c = co - q * C;
    int dh = (q == 1 || q == 3) ? 1 : 0;
    int dw = (q >= 2) ? 1 : 0;
    y[i] = x[(((size_t)b * C + c) * H + 2 * ho + dh) * W + 2 * wo + dw];
}

// inverse pixel-merge: (B,4C,H,W) -> (B,C,2H,2W)
__global__ void k_p2i(const float* __restrict__ x, float* __restrict__ y,
                      int Bn, int C, int H, int W) {
    int Ho = 2 * H, Wo = 2 * W;
    int i = blockIdx.x * blockDim.x + threadIdx.x;
    int total = Bn * C * Ho * Wo;
    if (i >= total) return;
    int ow = i % Wo, oh = (i / Wo) % Ho, c = (i / (Wo * Ho)) % C, b = i / (Wo * Ho * C);
    int h = oh >> 1, dh = oh & 1, w = ow >> 1, dw = ow & 1;
    int q = dh + 2 * dw;
    y[i] = x[(((size_t)b * 4 * C + q * C + c) * H + h) * W + w];
}

// bilinear x2 (reference's interp), into channel window of (Bn,dstC,2H,2W)
__global__ void k_up2(const float* __restrict__ x, float* __restrict__ y,
                      int Bn, int C, int H, int W, int dstC, int c0) {
    int Ho = 2 * H, Wo = 2 * W;
    int i = blockIdx.x * blockDim.x + threadIdx.x;
    int total = Bn * C * Ho * Wo;
    if (i >= total) return;
    int ow = i % Wo, oh = (i / Wo) % Ho, c = (i / (Wo * Ho)) % C, b = i / (Wo * Ho * C);
    float shf = (float)(H - 1) / (float)(Ho - 1);
    float swf = (float)(W - 1) / (float)(Wo - 1);
    float phf = oh * shf, pwf = ow * swf;
    int h0 = (int)floorf(phf); int h1 = min(h0 + 1, H - 1); float fh = phf - h0;
    int w0 = (int)floorf(pwf); int w1 = min(w0 + 1, W - 1); float fw = pwf - w0;
    const float* xp = x + ((size_t)b * C + c) * H * W;
    float v00 = xp[h0 * W + w0], v01 = xp[h0 * W + w1];
    float v10 = xp[h1 * W + w0], v11 = xp[h1 * W + w1];
    float v = (v00 * (1.f - fw) + v01 * fw) * (1.f - fh)
            + (v10 * (1.f - fw) + v11 * fw) * fh;
    y[(((size_t)b * dstC + c0 + c) * Ho + oh) * Wo + ow] = v;
}

__global__ void k_copy_chan(const float* __restrict__ x, float* __restrict__ y,
                            int Bn, int C, int H, int W, int dstC, int c0) {
    int i = blockIdx.x * blockDim.x + threadIdx.x;
    int total = Bn * C * H * W;
    if (i >= total) return;
    int w = i % W, h = (i / W) % H, c = (i / (W * H)) % C, b = i / (W * H * C);
    y[(((size_t)b * dstC + c0 + c) * H + h) * W + w] = x[i];
}

__global__ void k_add(float* __restrict__ dst, const float* __restrict__ src, int n) {
    int i = blockIdx.x * blockDim.x + threadIdx.x;
    if (i < n) dst[i] += src[i];
}

// row softmax, one 256-thread block per row of length L
__global__ void k_softmax(float* __restrict__ S, int L) {
    __shared__ float red[256];
    float* p = S + (size_t)blockIdx.x * L;
    int t = threadIdx.x;
    float mx = -3.4e38f;
    for (int i = t; i < L; i += 256) mx = fmaxf(mx, p[i]);
    red[t] = mx; __syncthreads();
    for (int s = 128; s > 0; s >>= 1) { if (t < s) red[t] = fmaxf(red[t], red[t + s]); __syncthreads(); }
    mx = red[0]; __syncthreads();
    float sum = 0.f;
    for (int i = t; i < L; i += 256) { float e = __expf(p[i] - mx); p[i] = e; sum += e; }
    red[t] = sum; __syncthreads();
    for (int s = 128; s > 0; s >>= 1) { if (t < s) red[t] += red[t + s]; __syncthreads(); }
    float inv = 1.f / red[0];
    for (int i = t; i < L; i += 256) p[i] *= inv;
}

// ---------------------------------------------------------------------------
extern "C" void kernel_launch(void* const* d_in, const int* in_sizes, int n_in,
                              void* d_out, int out_size, void* d_ws, size_t ws_size,
                              hipStream_t stream) {
    (void)in_sizes; (void)n_in; (void)out_size; (void)ws_size;
    const int B = 4;

    // -------- workspace layout (floats) --------
    float* ws = (float*)d_ws;
    float* X1 = ws;                      // 4x 64x256x256 = 16,777,216
    float* X2 = X1 + 16777216;           // 4x128x128x128 =  8,388,608
    float* X3 = X2 + 8388608;            // 4x256x 64x 64 =  4,194,304
    float* X4 = X3 + 4194304;            // 4x512x 32x 32 =  2,097,152
    float* X5 = X4 + 2097152;            // 4x512x 16x 16 =    524,288
    float* T0 = X5 + 524288;             // 16,777,216 each; T0..T1 contiguous
    float* T1 = T0 + 16777216;           //   (spanned as the 33.5M up4 concat)
    float* T2 = T1 + 16777216;
    float* T3 = T2 + 16777216;
    float* SB = T3 + 16777216;           // softmax buf, max 4x512x512 = 1,048,576
    float* SS = SB + 1048576;            // per-conv scale/shift arena
    _Float16* W16 = (_Float16*)(SS + 16384);  // f16 weight arena (~15M halves)

    float*    sscur = SS;
    _Float16* wcur  = W16;

    int ix = 0;
    auto nxt = [&]() { return (const float*)d_in[ix++]; };
    auto eltN = [](int n) { return (n + 255) / 256; };
    auto padTo = [](int v, int a) { return (v + a - 1) & ~(a - 1); };

    // repack conv weight into padded f16 arena; returns base pointer
    auto repack = [&](const float* src, int Cout, int Cin, int ks) -> const _Float16* {
        int CoutP = padTo(Cout, 16), CinP = padTo(Cin, 32);
        int n = ks * ks * CoutP * CinP;
        _Float16* d = wcur; wcur += n;
        k_repack_w<<<eltN(n), 256, 0, stream>>>(src, d, Cout, Cin, ks, CoutP, CinP);
        return d;
    };
    auto bnp = [&](const float* g, const float* bb, const float* m, const float* v,
                   const float* cb, int C) -> float* {
        float* s = sscur; sscur += 2 * C;
        k_bnprep<<<eltN(C), 256, 0, stream>>>(g, bb, m, v, cb, s, s + C, C);
        return s;
    };

    auto conv = [&](const float* x, const _Float16* w, const float* scale,
                    const float* shift, float* out, int Cin, int H, int W,
                    int Cout, int ks, int pad, int relu) {
        int CoutP = padTo(Cout, 16), CinP = padTo(Cin, 32);
        if (CoutP % 64 == 0) {
            int waves = B * (CoutP / 64) * ((H * W) / 16);
            k_conv_wmma<4><<<(waves + 7) / 8, 256, 0, stream>>>(
                x, w, scale, shift, out, B, Cin, H, W, Cout, ks, pad, relu, CinP, CoutP);
        } else {
            int waves = B * (CoutP / 16) * ((H * W) / 16);
            k_conv_wmma<1><<<(waves + 7) / 8, 256, 0, stream>>>(
                x, w, scale, shift, out, B, Cin, H, W, Cout, ks, pad, relu, CinP, CoutP);
        }
    };
    auto gemm = [&](const float* A, const float* Bm, float* C, int M, int N, int K,
                    int lda, int ldb, int ldc, int ta, int tb,
                    long long sA, long long sB, long long sC) {
        int waves = B * (M / 64) * (N / 16);
        int blocks = (waves + 7) / 8;
        if (ta == 0 && tb == 0)
            k_gemm_wmma<0, 0><<<blocks, 256, 0, stream>>>(A, Bm, C, M, N, K, lda, ldb, ldc, sA, sB, sC, B);
        else if (ta == 0 && tb == 1)
            k_gemm_wmma<0, 1><<<blocks, 256, 0, stream>>>(A, Bm, C, M, N, K, lda, ldb, ldc, sA, sB, sC, B);
        else
            k_gemm_wmma<1, 0><<<blocks, 256, 0, stream>>>(A, Bm, C, M, N, K, lda, ldb, ldc, sA, sB, sC, B);
    };
    auto dw = [&](const float* x, const float* w, const float* bias, float* out,
                  int C, int H, int W) {
        int n = B * C * H * W;
        k_dwconv3<<<eltN(n), 256, 0, stream>>>(x, w, bias, out, B, C, H, W);
    };
    auto pool = [&](const float* x, float* y, int C, int H, int W) {
        int n = B * C * (H / 2) * (W / 2);
        k_pool2<<<eltN(n), 256, 0, stream>>>(x, y, B, C, H, W);
    };
    auto pmk = [&](const float* x, float* y, int C, int H, int W) {
        int n = B * 4 * C * (H / 2) * (W / 2);
        k_pm<<<eltN(n), 256, 0, stream>>>(x, y, B, C, H, W);
    };
    auto p2ik = [&](const float* x, float* y, int C, int H, int W) {
        int n = B * C * 2 * H * 2 * W;
        k_p2i<<<eltN(n), 256, 0, stream>>>(x, y, B, C, H, W);
    };
    auto up2k = [&](const float* x, float* y, int C, int H, int W, int dstC, int c0) {
        int n = B * C * 2 * H * 2 * W;
        k_up2<<<eltN(n), 256, 0, stream>>>(x, y, B, C, H, W, dstC, c0);
    };
    auto copyc = [&](const float* x, float* y, int C, int H, int W, int dstC, int c0) {
        int n = B * C * H * W;
        k_copy_chan<<<eltN(n), 256, 0, stream>>>(x, y, B, C, H, W, dstC, c0);
    };
    auto addk = [&](float* dst, const float* src, int n) {
        k_add<<<eltN(n), 256, 0, stream>>>(dst, src, n);
    };
    auto softmax = [&](float* S, int rows, int L) {
        k_softmax<<<rows, 256, 0, stream>>>(S, L);
    };

    // -------- parameter loading (setup_inputs insertion order) --------
    struct DCP { const _Float16* w1; const float *sc1, *sh1;
                 const _Float16* w2; const float *sc2, *sh2; };
    auto loadDC = [&](int Cin, int Cout) -> DCP {
        const float* w1 = nxt(); const float* b1 = nxt();
        const float* g1 = nxt(); const float* be1 = nxt();
        const float* m1 = nxt(); const float* v1 = nxt();
        const float* w2 = nxt(); const float* b2 = nxt();
        const float* g2 = nxt(); const float* be2 = nxt();
        const float* m2 = nxt(); const float* v2 = nxt();
        DCP p;
        p.w1 = repack(w1, Cout, Cin, 3);
        float* s1 = bnp(g1, be1, m1, v1, b1, Cout); p.sc1 = s1; p.sh1 = s1 + Cout;
        p.w2 = repack(w2, Cout, Cout, 3);
        float* s2 = bnp(g2, be2, m2, v2, b2, Cout); p.sc2 = s2; p.sh2 = s2 + Cout;
        return p;
    };
    struct PMP { const float *wq, *bq, *wk, *bk; const _Float16* wc; };
    auto loadPMP = [&](int C) -> PMP {
        PMP p;
        p.wq = nxt(); p.bq = nxt(); p.wk = nxt(); p.bk = nxt();
        p.wc = repack(nxt(), C, C, 1);
        return p;
    };

    const float* Xin = nxt();                 // 'x' (4,3,256,256)
    DCP inc  = loadDC(3, 64);
    PMP pm1  = loadPMP(64);
    DCP dn1  = loadDC(64, 128);
    PMP pm2  = loadPMP(128);
    DCP dn2  = loadDC(128, 256);
    PMP pm3  = loadPMP(256);
    DCP dn3  = loadDC(256, 512);
    PMP pm4  = loadPMP(512);
    DCP dn4  = loadDC(512, 512);
    const float* cawq = nxt(); const float* cabq = nxt();
    const float* cawk = nxt(); const float* cabk = nxt();
    DCP up1  = loadDC(1024, 256);
    DCP up2p = loadDC(512, 128);
    DCP up3p = loadDC(256, 64);
    DCP up4p = loadDC(128, 64);
    const _Float16* outw = repack(nxt(), 1, 64, 1);
    const float*    outb = nxt();

    // -------- building blocks --------
    auto runDC = [&](const float* x, int Cin, int H, int W, int Cout,
                     const DCP& p, float* tmp, float* out) {
        conv(x,   p.w1, p.sc1, p.sh1, tmp, Cin,  H, W, Cout, 3, 1, 1);
        conv(tmp, p.w2, p.sc2, p.sh2, out, Cout, H, W, Cout, 3, 1, 1);
    };

    auto runPMPA = [&](float* x, int C, int H, int W, int nm, const PMP& p) {
        // Q = pm^n(dw(x,wq)+bq)  (ping T0<->T1)
        dw(x, p.wq, p.bq, T0, C, H, W);
        float *qa = T0, *qb = T1; int c = C, h = H, w = W;
        for (int i = 0; i < nm; ++i) { pmk(qa, qb, c, h, w); float* t = qa; qa = qb; qb = t; c <<= 2; h >>= 1; w >>= 1; }
        float *Q = qa, *qf = qb;
        // K (ping T2<->T3)
        dw(x, p.wk, p.bk, T2, C, H, W);
        float *ka = T2, *kb = T3; c = C; h = H; w = W;
        for (int i = 0; i < nm; ++i) { pmk(ka, kb, c, h, w); float* t = ka; ka = kb; kb = t; c <<= 2; h >>= 1; w >>= 1; }
        float *Kp = ka, *kf = kb;
        // V = pm^n(x) (ping between the two free slots)
        const float* cur = x; float *va = qf, *vb = kf; c = C; h = H; w = W;
        for (int i = 0; i < nm; ++i) { pmk(cur, va, c, h, w); cur = va; float* t = va; va = vb; vb = t; c <<= 2; h >>= 1; w >>= 1; }
        const float* V = cur;
        int Cm = c, hw = h * w;              // hw == 256 for every stage
        // S[i,j] = sum_c Q[c,i]K[c,j]  (A strided, B strided) -> softmax rows
        gemm(Q, Kp, SB, hw, hw, Cm, hw, hw, hw, 1, 0,
             (long long)Cm * hw, (long long)Cm * hw, (long long)hw * hw);
        softmax(SB, B * hw, hw);
        // E[c,i] = sum_j V[c,j] S[i,j]  (A contiguous, B contiguous-K)
        float* E = Q;
        gemm(V, SB, E, Cm, hw, hw, hw, hw, hw, 0, 1,
             (long long)Cm * hw, (long long)hw * hw, (long long)Cm * hw);
        // unmerge (ping E-slot <-> K-slot), then +x, 1x1 conv, residual into x
        float *ea = E, *eb = Kp;
        c = Cm; h = H >> nm; w = W >> nm;
        for (int i = 0; i < nm; ++i) { p2ik(ea, eb, c >> 2, h, w); float* t = ea; ea = eb; eb = t; c >>= 2; h <<= 1; w <<= 1; }
        addk(ea, x, B * C * H * W);                         // E + x
        conv(ea, p.wc, nullptr, nullptr, eb, C, H, W, C, 1, 0, 0);
        addk(x, eb, B * C * H * W);                         // x += pmpa(x)
    };

    auto runCA = [&](float* x, int C, int H, int W) {
        int hw = H * W;                                     // 256
        dw(x, cawq, cabq, T0, C, H, W);
        dw(x, cawk, cabk, T1, C, H, W);
        // S[i,j] = sum_k Bq[i,k] Ck[j,k]  (both contiguous-K)
        gemm(T0, T1, SB, C, C, hw, hw, hw, C, 0, 1,
             (long long)C * hw, (long long)C * hw, (long long)C * C);
        softmax(SB, B * C, C);
        // E[i,k] = sum_j S[i,j] x[j,k]  (A contiguous, B strided)
        gemm(SB, x, T2, C, hw, C, C, hw, hw, 0, 0,
             (long long)C * C, (long long)C * hw, (long long)C * hw);
        addk(x, T2, B * C * hw);                            // E + x
    };

    // ------------------------------ forward ------------------------------
    runDC(Xin, 3, 256, 256, 64, inc, T0, X1);
    runPMPA(X1, 64, 256, 256, 4, pm1);

    pool(X1, T0, 64, 256, 256);
    runDC(T0, 64, 128, 128, 128, dn1, T1, X2);
    runPMPA(X2, 128, 128, 128, 3, pm2);

    pool(X2, T0, 128, 128, 128);
    runDC(T0, 128, 64, 64, 256, dn2, T1, X3);
    runPMPA(X3, 256, 64, 64, 2, pm3);

    pool(X3, T0, 256, 64, 64);
    runDC(T0, 256, 32, 32, 512, dn3, T1, X4);
    runPMPA(X4, 512, 32, 32, 1, pm4);

    pool(X4, T0, 512, 32, 32);
    runDC(T0, 512, 16, 16, 512, dn4, T1, X5);
    runCA(X5, 512, 16, 16);

    // up1: concat(x4, up2(x5)) in T0 (4,1024,32,32) -> y in T2 (4,256,32,32)
    up2k(X5, T0, 512, 16, 16, 1024, 512);
    copyc(X4, T0, 512, 32, 32, 1024, 0);
    runDC(T0, 1024, 32, 32, 256, up1, T1, T2);

    // up2: concat(x3, up2(y)) in T0 (4,512,64,64) -> y in T2 (4,128,64,64)
    up2k(T2, T0, 256, 32, 32, 512, 256);
    copyc(X3, T0, 256, 64, 64, 512, 0);
    runDC(T0, 512, 64, 64, 128, up2p, T1, T2);

    // up3: concat(x2, up2(y)) in T0 (4,256,128,128) -> y in T2 (4,64,128,128)
    up2k(T2, T0, 128, 64, 64, 256, 128);
    copyc(X2, T0, 128, 128, 128, 256, 0);
    runDC(T0, 256, 128, 128, 64, up3p, T1, T2);

    // up4: concat(x1, up2(y)) spans T0+T1 (4,128,256,256) -> y in T3
    up2k(T2, T0, 64, 128, 128, 128, 64);
    copyc(X1, T0, 64, 256, 256, 128, 0);
    runDC(T0, 128, 256, 256, 64, up4p, T2, T3);

    // final 1x1 conv, bias only
    conv(T3, outw, nullptr, outb, (float*)d_out, 64, 256, 256, 1, 1, 0, 0);
}
